// Conv4Pim_group_arr_v2_10445360464136
// MI455X (gfx1250) — matrix operand order repaired and checked by reading, not compiled, not run on hardware
//
#include <hip/hip_runtime.h>

typedef __attribute__((ext_vector_type(16))) _Float16 v16h;
typedef __attribute__((ext_vector_type(8)))  _Float16 v8h;
typedef __attribute__((ext_vector_type(8)))  float    v8f;

// ---------------- problem config ----------------
constexpr int kIC   = 112;
constexpr int kOC   = 512;
constexpr int kH    = 56;
constexpr int kW    = 56;
constexpr int kN    = 16;
constexpr int kHW   = kH * kW;          // 3136
constexpr int kMTot = kN * kHW;         // 50176
// GEMM tiling
constexpr int BM = 128;
constexpr int BN = 128;
constexpr int NCHUNK = 36;              // 9 kernel positions * 4 ic-chunks of 32 (IC padded to 128)
constexpr int LSTR = 40;                // LDS row stride in halves (padded, conflict-free)
constexpr int THREADS = 512;            // 16 waves of 32
constexpr int kBufHalves = (BM + 2 * BN) * LSTR;   // 15360 halves per stage
constexpr int kBufBytes  = kBufHalves * 2;         // 30720 B
// prep layout: wq[branch][p(9)][c(4)][oc(512)][ch(32)]  (f16)
constexpr int kChunkHalves  = kOC * 32;            // 16384 halves per (b,p,c)
constexpr int kBranchHalves = 9 * 4 * kChunkHalves;
constexpr int kPrepTotal    = 2 * kBranchHalves;   // 1,179,648 halves = 2.36 MB

// ---------------- kernel 1: fold LSQ + 2-bit slice + scale into f16 weights ----------------
__global__ void prep_weights(const float* __restrict__ w,
                             const float* __restrict__ sw_p,
                             const float* __restrict__ sw_n,
                             _Float16* __restrict__ wq)
{
    int idx = blockIdx.x * blockDim.x + threadIdx.x;
    if (idx >= kPrepTotal) return;
    int ch = idx & 31;
    int o  = (idx >> 5) & 511;
    int c  = (idx >> 14) & 3;
    int bp = idx >> 16;            // = b*9 + p
    int p  = bp % 9;
    int b  = bp / 9;
    int i  = c * 32 + ch;
    float val = 0.f;
    if (i < kIC) {
        int kh  = p / 3, kw = p - 3 * (p / 3);
        int row = o >> 8;          // o / 256
        int col = i / 28;
        float s  = (b == 0 ? sw_p : sw_n)[row * 4 + col];
        float wv = w[((o * kIC + i) * 3 + kh) * 3 + kw];
        float ws = (b == 0) ? fmaxf(wv, 0.f) : fmaxf(-wv, 0.f);
        float lvl = rintf(fminf(fmaxf(ws / s, 0.f), 15.f)); // LSQ: clip then round (RNE)
        int sl = (((int)lvl) >> 2) & 3;                      // idx=1, split_bit=2 slice
        val = (float)sl * s;
    }
    wq[idx] = (_Float16)val;
}

// ---------------- kernel 2: implicit-GEMM conv, dual-branch WMMA ----------------
__global__ __launch_bounds__(THREADS)
void conv_wmma(const float* __restrict__ x,
               const _Float16* __restrict__ wq,
               const float* __restrict__ sp_p_ptr,
               const float* __restrict__ sp_n_ptr,
               float* __restrict__ out)
{
    // Per stage: A rows [0,128), Bp rows [128,256), Bn rows [256,384); stride 40 halves
    __shared__ __align__(16) _Float16 lds[2][kBufHalves]; // 61440 B

    const int tid  = threadIdx.x;
    const int lane = tid & 31;
    const int wave = tid >> 5;          // 0..15
    const int wm   = wave >> 2;         // 0..3 (M)
    const int wn   = wave & 3;          // 0..3 (N)

    const int bn = blockIdx.x & 3;      // 4 N-tiles
    const int bm = blockIdx.x >> 2;     // 392 M-tiles

    const unsigned ldsBase = (unsigned)(uintptr_t)&lds[0][0]; // LDS byte offset (addr[31:0])

    // ----- A global-load mapping: element (m, ch); m = lane + 32j (j<4), ch = (tid>>5) + 16l (l<2)
    int mrow[4], ohv[4], owv[4];
    int abase[4];
    #pragma unroll
    for (int j = 0; j < 4; ++j) {
        int m  = (tid & 31) + 32 * j;
        int mg = bm * BM + m;
        int n  = mg / kHW;
        int r  = mg - n * kHW;
        int oh = r / kW;
        int ow = r - oh * kW;
        mrow[j] = m; ohv[j] = oh; owv[j] = ow;
        abase[j] = n * kIC * kHW + oh * kW + ow;
    }
    const int chb = tid >> 5;           // 0..15

    // ----- B async-load mapping: 1 b128 (8 halves) per thread per branch per chunk
    const int b_oc = tid >> 2;          // 0..127
    const int b_ui = tid & 3;           // which 8-half group of the 32-ch row
    const int b_voff = ((bn * 128 + b_oc) * 4 + b_ui) * 16;  // byte offset inside chunk

    const float sp_p = *sp_p_ptr;
    const float sp_n = *sp_n_ptr;

    v8f acc[2][2][2] = {};              // [branch][mf][nf]
    float areg[8];

    auto issue_async_b = [&](int t, int buf) {
        const int p = t >> 2, c = t & 3;
        const int cb = (p * 4 + c) * kChunkHalves;
        #pragma unroll
        for (int b = 0; b < 2; ++b) {
            const _Float16* sa = wq + b * kBranchHalves + cb;   // uniform -> SGPR pair
            int laddr = (int)ldsBase + buf * kBufBytes
                      + ((BM + b * BN + b_oc) * LSTR + b_ui * 8) * 2;
            asm volatile("global_load_async_to_lds_b128 %0, %1, %2"
                         :: "v"(laddr), "v"(b_voff), "s"(sa) : "memory");
        }
    };

    auto wait_async = [&]() {
#if __has_builtin(__builtin_amdgcn_s_wait_asynccnt)
        __builtin_amdgcn_s_wait_asynccnt(0);
#else
        asm volatile("s_wait_asynccnt 0x0" ::: "memory");
#endif
    };

    auto global_load_a = [&](int t) {
        const int p  = t >> 2, c = t & 3;
        const int kh = p / 3, kw = p - 3 * kh;
        const int dh = kh - 1, dw = kw - 1;
        const int icb = c * 32;
        #pragma unroll
        for (int j = 0; j < 4; ++j) {
            int ih = ohv[j] + dh, iw = owv[j] + dw;
            bool inb = ((unsigned)ih < (unsigned)kH) && ((unsigned)iw < (unsigned)kW);
            #pragma unroll
            for (int l = 0; l < 2; ++l) {
                int ic = icb + chb + 16 * l;
                float v = 0.f;
                if (inb && ic < kIC)
                    v = x[abase[j] + ic * kHW + dh * kW + dw];
                areg[j * 2 + l] = v;
            }
        }
    };

    auto lds_store_a = [&](int buf) {
        #pragma unroll
        for (int j = 0; j < 4; ++j)
            #pragma unroll
            for (int l = 0; l < 2; ++l)
                lds[buf][mrow[j] * LSTR + chb + 16 * l] = (_Float16)areg[j * 2 + l];
    };

    const int r16 = lane & 15;
    const int kbA = (lane < 16) ? 0 : 16;   // A 16x32 f16: K {0-15} / {16-31} (contiguous)
    const int kbB = (lane < 16) ? 0 : 16;   // B 32x16 f16: K {0-15} / {16-31} (contiguous)

    auto compute = [&](int buf) {
        v16h afr[2], bfr[2][2];
        #pragma unroll
        for (int mf = 0; mf < 2; ++mf) {
            int row = wm * 32 + mf * 16 + r16;
            v8h lo = *(const v8h*)&lds[buf][row * LSTR + kbA];
            v8h hi = *(const v8h*)&lds[buf][row * LSTR + kbA + 8];
            afr[mf] = __builtin_shufflevector(lo, hi, 0,1,2,3,4,5,6,7,8,9,10,11,12,13,14,15);
        }
        #pragma unroll
        for (int b = 0; b < 2; ++b)
            #pragma unroll
            for (int nf = 0; nf < 2; ++nf) {
                int row = BM + b * BN + wn * 32 + nf * 16 + r16;
                v8h lo = *(const v8h*)&lds[buf][row * LSTR + kbB];
                v8h hi = *(const v8h*)&lds[buf][row * LSTR + kbB + 8];
                bfr[b][nf] = __builtin_shufflevector(lo, hi, 0,1,2,3,4,5,6,7,8,9,10,11,12,13,14,15);
            }
        #pragma unroll
        for (int b = 0; b < 2; ++b)
            #pragma unroll
            for (int mf = 0; mf < 2; ++mf)
                #pragma unroll
                for (int nf = 0; nf < 2; ++nf)
                    acc[b][mf][nf] = __builtin_amdgcn_wmma_f32_16x16x32_f16(
                        false, afr[mf], false, bfr[b][nf],
                        (short)0, acc[b][mf][nf], false, false);
    };

    // ----- software-pipelined main loop (double-buffered LDS, async B copies) -----
    issue_async_b(0, 0);
    global_load_a(0);
    lds_store_a(0);
    wait_async();
    __syncthreads();
    for (int t = 0; t < NCHUNK; ++t) {
        const int cur = t & 1;
        const int nxt = cur ^ 1;
        if (t + 1 < NCHUNK) {
            issue_async_b(t + 1, nxt);   // global -> LDS, ASYNCcnt-tracked
            global_load_a(t + 1);
        }
        compute(cur);
        if (t + 1 < NCHUNK) lds_store_a(nxt);
        wait_async();
        __syncthreads();
    }

    // ----- epilogue: psum LSQ quantize both branches, subtract, coalesce via LDS -----
    // Two passes of 64 oc x 128 m, staged as [oc][m] f32 with padded stride 132.
    float* lfs = (float*)&lds[0][0];    // 64*132*4 = 33792 B, reuses main LDS
    const int n_l  = lane & 15;
    const int m_hi = (lane >> 4) * 8;
    #pragma unroll
    for (int pass = 0; pass < 2; ++pass) {
        __syncthreads();
        if ((wn >> 1) == pass) {
            #pragma unroll
            for (int mf = 0; mf < 2; ++mf)
                #pragma unroll
                for (int nf = 0; nf < 2; ++nf)
                    #pragma unroll
                    for (int r = 0; r < 8; ++r) {
                        int oc_l = (wn & 1) * 32 + nf * 16 + n_l;   // 0..63
                        int m_l  = wm * 32 + mf * 16 + r + m_hi;    // 0..127
                        float pv = acc[0][mf][nf][r];
                        float nv = acc[1][mf][nf][r];
                        float qp = rintf(fminf(fmaxf(pv / sp_p, -128.f), 127.f)) * sp_p;
                        float qn = rintf(fminf(fmaxf(nv / sp_n, -128.f), 127.f)) * sp_n;
                        lfs[oc_l * 132 + m_l] = qp - qn;
                    }
        }
        __syncthreads();
        // 8192 elems, lane-contiguous in m -> coalesced global stores
        #pragma unroll
        for (int i = 0; i < 16; ++i) {
            int e    = tid + THREADS * i;
            int oc_l = e >> 7;
            int m    = e & 127;
            int oc   = bn * BN + pass * 64 + oc_l;
            int mg   = bm * BM + m;
            int n    = mg / kHW;
            int rr   = mg - n * kHW;
            out[(n * kOC + oc) * kHW + rr] = lfs[oc_l * 132 + m];
        }
    }
}

// ---------------- launch ----------------
extern "C" void kernel_launch(void* const* d_in, const int* in_sizes, int n_in,
                              void* d_out, int out_size, void* d_ws, size_t ws_size,
                              hipStream_t stream)
{
    const float* x    = (const float*)d_in[0];
    const float* w    = (const float*)d_in[1];
    const float* sw_p = (const float*)d_in[2];
    const float* sw_n = (const float*)d_in[3];
    const float* sp_p = (const float*)d_in[4];
    const float* sp_n = (const float*)d_in[5];

    _Float16* wq = (_Float16*)d_ws;   // 2,359,296 bytes required

    prep_weights<<<(kPrepTotal + 255) / 256, 256, 0, stream>>>(w, sw_p, sw_n, wq);

    dim3 grid((kMTot / BM) * (kOC / BN));   // 392 * 4 = 1568 blocks
    conv_wmma<<<grid, THREADS, 0, stream>>>(x, wq, sp_p, sp_n, (float*)d_out);
}
